// Voformer_6914897347082
// MI455X (gfx1250) — matrix operand order repaired and checked by reading, not compiled, not run on hardware
//
#include <hip/hip_runtime.h>
#include <math.h>

// ---------------------------------------------------------------------------
// Types for CDNA5 WMMA (gfx1250, wave32)
// ---------------------------------------------------------------------------
typedef __attribute__((ext_vector_type(16))) __bf16    bf16x16;
typedef __attribute__((ext_vector_type(8)))  float     floatx8;
typedef __attribute__((ext_vector_type(4)))  unsigned  uintx4;

union Frag16 {
    bf16x16        v;
    uintx4         q[2];
    unsigned short s[16];
};

__device__ __forceinline__ unsigned short f32_to_bf16(float f) {
    union { float f; unsigned u; } x;
    x.f = f;
    unsigned u = x.u;
    unsigned r = u + 0x7FFFu + ((u >> 16) & 1u);   // round-to-nearest-even
    return (unsigned short)(r >> 16);
}

#define BSZ    8
#define LSEQ   1024
#define DMODEL 256
#define DFF    1024
#define TOPK   256
#define NROWS  (BSZ * LSEQ)   // 8192

// ---------------------------------------------------------------------------
// Weight convert: W[K,N] f32 -> WT[N,K] bf16  (transposed so WMMA B-fragments
// are contiguous 16B loads)
// ---------------------------------------------------------------------------
__global__ void wconv_kernel(const float* __restrict__ W,
                             unsigned short* __restrict__ WT, int K, int N) {
    int idx = blockIdx.x * blockDim.x + threadIdx.x;
    if (idx >= K * N) return;
    int n = idx / K, k = idx % K;
    WT[idx] = f32_to_bf16(W[(size_t)k * N + n]);
}

// ---------------------------------------------------------------------------
// Input embedding: h = x @ in_w + in_b + sinusoidal PE
// ---------------------------------------------------------------------------
__global__ void embed_kernel(const float* __restrict__ x,
                             const float* __restrict__ in_w,
                             const float* __restrict__ in_b,
                             float* __restrict__ h) {
    int idx = blockIdx.x * blockDim.x + threadIdx.x;   // B*L*256
    int d = idx & 255;
    int l = (idx >> 8) & 1023;
    int b = idx >> 18;
    float x0 = x[(size_t)(b * LSEQ + l) * 2 + 0];
    float x1 = x[(size_t)(b * LSEQ + l) * 2 + 1];
    int   i  = d >> 1;
    float dv = __expf((float)(2 * i) * (-9.210340371976184f / 256.0f));
    float ang = (float)l * dv;
    float pe  = (d & 1) ? __cosf(ang) : __sinf(ang);
    h[idx] = x0 * in_w[d] + x1 * in_w[DMODEL + d] + in_b[d] + pe;
}

// ---------------------------------------------------------------------------
// Flexible LayerNorm over last dim (256). One wave per row, 8 waves/block.
//   y = LN(X (+ Xadd)) * g + b ; optionally write f32, bf16, or accumulate.
// ---------------------------------------------------------------------------
__global__ __launch_bounds__(256) void ln_kernel(
        const float* __restrict__ X, const float* __restrict__ Xadd,
        const float* __restrict__ g, const float* __restrict__ bta,
        float* __restrict__ Yf, unsigned short* __restrict__ Yb,
        float* __restrict__ Yacc) {
    int wave = threadIdx.x >> 5, lane = threadIdx.x & 31;
    int row  = blockIdx.x * 8 + wave;
    const float* xr = X + (size_t)row * DMODEL;
    float vals[8];
    float s = 0.f, ss = 0.f;
#pragma unroll
    for (int j = 0; j < 8; ++j) {
        int c = lane + j * 32;
        float v = xr[c];
        if (Xadd) v += Xadd[(size_t)row * DMODEL + c];
        vals[j] = v;
        s += v; ss += v * v;
    }
#pragma unroll
    for (int m = 1; m < 32; m <<= 1) {
        s  += __shfl_xor(s,  m, 32);
        ss += __shfl_xor(ss, m, 32);
    }
    float mean = s * (1.0f / DMODEL);
    float var  = ss * (1.0f / DMODEL) - mean * mean;
    float rs   = rsqrtf(var + 1e-5f);
#pragma unroll
    for (int j = 0; j < 8; ++j) {
        int c = lane + j * 32;
        float y = (vals[j] - mean) * rs * g[c] + bta[c];
        size_t idx = (size_t)row * DMODEL + c;
        if (Yf)   Yf[idx] = y;
        if (Yb)   Yb[idx] = f32_to_bf16(y);
        if (Yacc) Yacc[idx] += y;
    }
}

// ---------------------------------------------------------------------------
// Generic bf16 GEMM via v_wmma_f32_16x16x32_bf16, 2x2 register blocking.
//   C[M,N] = A[M,K] @ WT[N,K]^T (+bias) (+add)
// One wave -> one 32x32 tile (4 WMMA accumulators); 8 waves per block.
// Each A/B fragment feeds 2 WMMAs -> 16 FLOP/byte from memory.
// ---------------------------------------------------------------------------
__global__ __launch_bounds__(256) void gemm_bf16_wave(
        const unsigned short* __restrict__ A,
        const unsigned short* __restrict__ WT,
        const float* __restrict__ bias,
        const float* __restrict__ add,
        float* __restrict__ Cf, unsigned short* __restrict__ Cb,
        int M, int N, int K) {
    int wave = threadIdx.x >> 5, lane = threadIdx.x & 31;
    int nblk = N >> 5;
    int tile = blockIdx.x * 8 + wave;
    int tm = tile / nblk, tn = tile % nblk;
    if (tm >= (M >> 5)) return;
    int m0 = tm << 5, n0 = tn << 5;

    int akb = (lane >> 4) << 3;             // A lane K base (0 or 8)
    int bkb = (lane >> 4) << 3;             // B lane K base (0 or 8)
    const unsigned short* pa0 = A  + (size_t)(m0 + (lane & 15)) * K + akb;
    const unsigned short* pa1 = pa0 + (size_t)16 * K;
    const unsigned short* pb0 = WT + (size_t)(n0 + (lane & 15)) * K + bkb;
    const unsigned short* pb1 = pb0 + (size_t)16 * K;

    floatx8 acc00 = {}, acc01 = {}, acc10 = {}, acc11 = {};
    for (int k0 = 0; k0 < K; k0 += 32) {
        __builtin_prefetch(pa0 + k0 + 128, 0, 1);
        __builtin_prefetch(pb0 + k0 + 128, 0, 1);
        Frag16 a0, a1, b0, b1;
        a0.q[0] = *(const uintx4*)(pa0 + k0);        // K = akb .. akb+7
        a0.q[1] = *(const uintx4*)(pa0 + k0 + 16);   // K = akb+16 .. akb+23
        a1.q[0] = *(const uintx4*)(pa1 + k0);
        a1.q[1] = *(const uintx4*)(pa1 + k0 + 16);
        b0.q[0] = *(const uintx4*)(pb0 + k0);        // K = bkb .. bkb+7
        b0.q[1] = *(const uintx4*)(pb0 + k0 + 16);   // K = bkb+16 .. bkb+23
        b1.q[0] = *(const uintx4*)(pb1 + k0);
        b1.q[1] = *(const uintx4*)(pb1 + k0 + 16);
        acc00 = __builtin_amdgcn_wmma_f32_16x16x32_bf16(
                    false, a0.v, false, b0.v, (short)0, acc00, false, false);
        acc01 = __builtin_amdgcn_wmma_f32_16x16x32_bf16(
                    false, a0.v, false, b1.v, (short)0, acc01, false, false);
        acc10 = __builtin_amdgcn_wmma_f32_16x16x32_bf16(
                    false, a1.v, false, b0.v, (short)0, acc10, false, false);
        acc11 = __builtin_amdgcn_wmma_f32_16x16x32_bf16(
                    false, a1.v, false, b1.v, (short)0, acc11, false, false);
    }

    auto epi = [&](floatx8 acc, int mi, int ni) {
        int n = n0 + ni * 16 + (lane & 15);
        int mbase = m0 + mi * 16 + ((lane >> 4) << 3);
        float bv = bias ? bias[n] : 0.f;
#pragma unroll
        for (int v = 0; v < 8; ++v) {
            size_t idx = (size_t)(mbase + v) * N + n;
            float r = acc[v] + bv + (add ? add[idx] : 0.f);
            if (Cf) Cf[idx] = r;
            if (Cb) Cb[idx] = f32_to_bf16(r);
        }
    };
    epi(acc00, 0, 0); epi(acc01, 0, 1);
    epi(acc10, 1, 0); epi(acc11, 1, 1);
}

// ---------------------------------------------------------------------------
// QKV GEMM: A=[8192,256]bf16, WT=[768,256]bf16, 2x2 register blocking.
// Scatters outputs into attention-friendly layouts:
//   q,k -> [BH, L, 32]; v -> [BH, 32, L] (transposed)
// ---------------------------------------------------------------------------
__global__ __launch_bounds__(256) void qkv_gemm_kernel(
        const unsigned short* __restrict__ A,
        const unsigned short* __restrict__ WT,
        const float* __restrict__ bias,
        unsigned short* __restrict__ qbuf,
        unsigned short* __restrict__ kbuf,
        unsigned short* __restrict__ vtbuf) {
    const int K = 256, N = 768;
    int wave = threadIdx.x >> 5, lane = threadIdx.x & 31;
    int nblk = N >> 5;                       // 24
    int tile = blockIdx.x * 8 + wave;
    int tm = tile / nblk, tn = tile % nblk;
    int m0 = tm << 5, n0 = tn << 5;

    int akb = (lane >> 4) << 3;
    int bkb = (lane >> 4) << 3;
    const unsigned short* pa0 = A  + (size_t)(m0 + (lane & 15)) * K + akb;
    const unsigned short* pa1 = pa0 + (size_t)16 * K;
    const unsigned short* pb0 = WT + (size_t)(n0 + (lane & 15)) * K + bkb;
    const unsigned short* pb1 = pb0 + (size_t)16 * K;

    floatx8 acc00 = {}, acc01 = {}, acc10 = {}, acc11 = {};
    for (int k0 = 0; k0 < K; k0 += 32) {
        __builtin_prefetch(pa0 + k0 + 128, 0, 1);
        __builtin_prefetch(pb0 + k0 + 128, 0, 1);
        Frag16 a0, a1, b0, b1;
        a0.q[0] = *(const uintx4*)(pa0 + k0);
        a0.q[1] = *(const uintx4*)(pa0 + k0 + 16);
        a1.q[0] = *(const uintx4*)(pa1 + k0);
        a1.q[1] = *(const uintx4*)(pa1 + k0 + 16);
        b0.q[0] = *(const uintx4*)(pb0 + k0);
        b0.q[1] = *(const uintx4*)(pb0 + k0 + 16);
        b1.q[0] = *(const uintx4*)(pb1 + k0);
        b1.q[1] = *(const uintx4*)(pb1 + k0 + 16);
        acc00 = __builtin_amdgcn_wmma_f32_16x16x32_bf16(
                    false, a0.v, false, b0.v, (short)0, acc00, false, false);
        acc01 = __builtin_amdgcn_wmma_f32_16x16x32_bf16(
                    false, a0.v, false, b1.v, (short)0, acc01, false, false);
        acc10 = __builtin_amdgcn_wmma_f32_16x16x32_bf16(
                    false, a1.v, false, b0.v, (short)0, acc10, false, false);
        acc11 = __builtin_amdgcn_wmma_f32_16x16x32_bf16(
                    false, a1.v, false, b1.v, (short)0, acc11, false, false);
    }

    auto epi = [&](floatx8 acc, int mi, int ni) {
        int n = n0 + ni * 16 + (lane & 15);
        int which = n >> 8, head = (n >> 5) & 7, d = n & 31;
        float bv = bias[n];
        int mbase = m0 + mi * 16 + ((lane >> 4) << 3);
#pragma unroll
        for (int v = 0; v < 8; ++v) {
            int m = mbase + v;
            int b = m >> 10, l = m & 1023;
            int bh = b * 8 + head;
            unsigned short val = f32_to_bf16(acc[v] + bv);
            if (which == 0)      qbuf [((size_t)(bh * LSEQ + l) << 5) + d] = val;
            else if (which == 1) kbuf [((size_t)(bh * LSEQ + l) << 5) + d] = val;
            else                 vtbuf[((size_t)(bh * 32 + d) << 10) + l] = val;
        }
    };
    epi(acc00, 0, 0); epi(acc01, 0, 1);
    epi(acc10, 1, 0); epi(acc11, 1, 1);
}

// ---------------------------------------------------------------------------
// ProbSparse attention. One block = one (b,h) and one 16-query tile.
// 128 threads (4 waves). LDS holds the full 16x1024 score row block.
//  1. scores = (Q Kt) * scale via WMMA      (one WMMA per 16-key tile)
//  2. per-row top-k(=256) threshold by 24-step value bisection (8 lanes/row)
//  3. sparse softmax written back to LDS
//  4. context = P @ V via WMMA (dk=32 -> 2 N-tiles), cross-wave LDS reduce
// ---------------------------------------------------------------------------
__global__ __launch_bounds__(128) void attn_kernel(
        const unsigned short* __restrict__ qb,
        const unsigned short* __restrict__ kb,
        const unsigned short* __restrict__ vt,
        unsigned short* __restrict__ ctxb) {
    __shared__ float sm[16 * LSEQ];           // 64 KB
    int bh    = blockIdx.x >> 6;
    int qtile = blockIdx.x & 63;
    int tid   = threadIdx.x;
    int wave  = tid >> 5, lane = tid & 31;
    const float scale = 0.17677669529663687f; // 1/sqrt(32)

    // --- Q fragment (A-layout), shared by all score WMMAs of this wave ---
    Frag16 qa;
    {
        int qrow  = qtile * 16 + (lane & 15);
        int kbase = (lane >> 4) << 3;
        const uintx4* p = (const uintx4*)(qb + ((size_t)(bh * LSEQ + qrow) << 5) + kbase);
        qa.q[0] = p[0];
        qa.q[1] = p[2];
    }

    // --- 1. scores into LDS ---
    for (int j = wave; j < 64; j += 4) {      // 64 key tiles of 16
        Frag16 kf;
        int key = j * 16 + (lane & 15);
        int kbb = (lane >> 4) << 3;
        const uintx4* p = (const uintx4*)(kb + ((size_t)(bh * LSEQ + key) << 5) + kbb);
        kf.q[0] = p[0];
        kf.q[1] = p[2];
        floatx8 s = {};
        s = __builtin_amdgcn_wmma_f32_16x16x32_bf16(
                false, qa.v, false, kf.v, (short)0, s, false, false);
        int n = lane & 15, mb = (lane >> 4) << 3;
#pragma unroll
        for (int v = 0; v < 8; ++v)
            sm[(mb + v) * LSEQ + j * 16 + n] = s[v] * scale;
    }
    __syncthreads();

    // --- 2+3. top-k threshold (bisection) + sparse softmax, 8 lanes/row ---
    {
        int r = tid >> 3, sub = tid & 7;
        float* row = sm + r * LSEQ + sub * 128;
        float mx = -3.4e38f, mn = 3.4e38f;
        for (int i = 0; i < 128; ++i) {
            float x = row[i];
            mx = fmaxf(mx, x);
            mn = fminf(mn, x);
        }
        for (int m = 1; m < 8; m <<= 1) {
            mx = fmaxf(mx, __shfl_xor(mx, m, 32));
            mn = fminf(mn, __shfl_xor(mn, m, 32));
        }
        float lo = mn, hi = mx;
        for (int it = 0; it < 24; ++it) {
            float mid = 0.5f * (lo + hi);
            int c = 0;
            for (int i = 0; i < 128; ++i) c += (row[i] >= mid) ? 1 : 0;
            for (int m = 1; m < 8; m <<= 1) c += __shfl_xor(c, m, 32);
            if (c >= TOPK) lo = mid; else hi = mid;
        }
        float thr = lo;
        float sum = 0.f;
        for (int i = 0; i < 128; ++i) {
            float x = row[i];
            if (x >= thr) sum += __expf(x - mx);
        }
        for (int m = 1; m < 8; m <<= 1) sum += __shfl_xor(sum, m, 32);
        float inv = 1.0f / sum;
        for (int i = 0; i < 128; ++i) {
            float x = row[i];
            row[i] = (x >= thr) ? __expf(x - mx) * inv : 0.f;
        }
    }
    __syncthreads();

    // --- 4. context = P @ V. Each wave: 8 key chunks of 32, both d-tiles ---
    floatx8 acc0 = {}, acc1 = {};
    for (int chunk = wave; chunk < 32; chunk += 4) {
        Frag16 pa;
        {
            int m = lane & 15;
            int kbase = (lane >> 4) << 3;
#pragma unroll
            for (int e = 0; e < 16; ++e) {
                int kl = (e < 8) ? (kbase + e) : (kbase + e + 8);
                pa.s[e] = f32_to_bf16(sm[m * LSEQ + chunk * 32 + kl]);
            }
        }
#pragma unroll
        for (int t = 0; t < 2; ++t) {
            Frag16 vb;
            int d   = t * 16 + (lane & 15);
            int kbb = (lane >> 4) << 3;
            const uintx4* p = (const uintx4*)(vt + ((size_t)(bh * 32 + d) << 10)
                                                 + chunk * 32 + kbb);
            vb.q[0] = p[0];
            vb.q[1] = p[2];
            if (t == 0)
                acc0 = __builtin_amdgcn_wmma_f32_16x16x32_bf16(
                           false, pa.v, false, vb.v, (short)0, acc0, false, false);
            else
                acc1 = __builtin_amdgcn_wmma_f32_16x16x32_bf16(
                           false, pa.v, false, vb.v, (short)0, acc1, false, false);
        }
    }
    __syncthreads();                       // prob rows no longer needed
    {
        int n = lane & 15, mb = (lane >> 4) << 3;
#pragma unroll
        for (int v = 0; v < 8; ++v) {
            sm[(wave * 2 + 0) * 256 + (mb + v) * 16 + n] = acc0[v];
            sm[(wave * 2 + 1) * 256 + (mb + v) * 16 + n] = acc1[v];
        }
    }
    __syncthreads();
    int b = bh >> 3, head = bh & 7;
    for (int o = tid; o < 512; o += 128) {
        int m = o >> 5, d = o & 31;
        int t = d >> 4, nn = d & 15;
        float s = 0.f;
#pragma unroll
        for (int w = 0; w < 4; ++w) s += sm[(w * 2 + t) * 256 + m * 16 + nn];
        int q = qtile * 16 + m;
        ctxb[((size_t)(b * LSEQ + q) << 8) + head * 32 + d] = f32_to_bf16(s);
    }
}

// ---------------------------------------------------------------------------
// Volatilite: per (batch, ff-column) std over the sequence axis, then f*=vol,
// written as bf16 for the ff2 GEMM. Block = 256 consecutive columns of one
// batch -> every row sweep is a fully coalesced contiguous 1KB access.
// ---------------------------------------------------------------------------
__global__ __launch_bounds__(256) void vol_kernel(const float* __restrict__ F,
                                                  unsigned short* __restrict__ Fb) {
    int b  = blockIdx.x >> 2;                 // batch
    int j  = (blockIdx.x & 3) * 256 + threadIdx.x;   // column 0..1023
    const float* base = F + ((size_t)b << 20) + j;   // [1024 rows][1024 cols]
    float s = 0.f, ss = 0.f;
    for (int l = 0; l < LSEQ; ++l) {
        float v = base[(size_t)l * DFF];
        s += v; ss += v * v;
    }
    float mean = s * (1.0f / LSEQ);
    float var  = ss * (1.0f / LSEQ) - mean * mean;
    float vol  = sqrtf(fmaxf(var, 0.f));
    unsigned short* ob = Fb + ((size_t)b << 20) + j;
    for (int l = 0; l < LSEQ; ++l) {
        size_t off = (size_t)l * DFF;
        ob[off] = f32_to_bf16(base[off] * vol);
    }
}

// ---------------------------------------------------------------------------
// Host orchestration
// ---------------------------------------------------------------------------
extern "C" void kernel_launch(void* const* d_in, const int* in_sizes, int n_in,
                              void* d_out, int out_size, void* d_ws, size_t ws_size,
                              hipStream_t stream) {
    (void)in_sizes; (void)n_in; (void)out_size; (void)ws_size;

    const float* x     = (const float*)d_in[0];
    const float* in_w  = (const float*)d_in[1];
    const float* in_b  = (const float*)d_in[2];
    const float* ln1_g = (const float*)d_in[3];
    const float* ln1_b = (const float*)d_in[4];
    const float* qkv_w = (const float*)d_in[5];
    const float* qkv_b = (const float*)d_in[6];
    const float* out_w = (const float*)d_in[7];
    const float* out_b = (const float*)d_in[8];
    const float* lna_g = (const float*)d_in[9];
    const float* lna_b = (const float*)d_in[10];
    const float* ln2_g = (const float*)d_in[11];
    const float* ln2_b = (const float*)d_in[12];
    const float* ff1_w = (const float*)d_in[13];
    const float* ff1_b = (const float*)d_in[14];
    const float* ff2_w = (const float*)d_in[15];
    const float* ff2_b = (const float*)d_in[16];
    const float* lnf_g = (const float*)d_in[17];
    const float* lnf_b = (const float*)d_in[18];

    char* wp = (char*)d_ws;
    auto alloc = [&](size_t bytes) -> void* {
        void* p = (void*)wp;
        wp += (bytes + 255) & ~(size_t)255;
        return p;
    };

    float*          h    = (float*)         alloc((size_t)NROWS * DMODEL * 4);
    float*          s2   = (float*)         alloc((size_t)NROWS * DMODEL * 4);
    unsigned short* s2b  = (unsigned short*)alloc((size_t)NROWS * DMODEL * 2);
    unsigned short* qbuf = (unsigned short*)alloc((size_t)64 * LSEQ * 32 * 2);
    unsigned short* kbuf = (unsigned short*)alloc((size_t)64 * LSEQ * 32 * 2);
    unsigned short* vtb  = (unsigned short*)alloc((size_t)64 * 32 * LSEQ * 2);
    unsigned short* ctxb = (unsigned short*)alloc((size_t)NROWS * DMODEL * 2);
    float*          oprj = (float*)         alloc((size_t)NROWS * DMODEL * 4);
    float*          ff   = (float*)         alloc((size_t)NROWS * DFF * 4);
    unsigned short* ffb  = (unsigned short*)alloc((size_t)NROWS * DFF * 2);
    unsigned short* WTq  = (unsigned short*)alloc((size_t)768 * 256 * 2);
    unsigned short* WTo  = (unsigned short*)alloc((size_t)256 * 256 * 2);
    unsigned short* WTf1 = (unsigned short*)alloc((size_t)1024 * 256 * 2);
    unsigned short* WTf2 = (unsigned short*)alloc((size_t)256 * 1024 * 2);

    // weights -> transposed bf16
    wconv_kernel<<<(256 * 768 + 255) / 256, 256, 0, stream>>>(qkv_w, WTq, 256, 768);
    wconv_kernel<<<(256 * 256 + 255) / 256, 256, 0, stream>>>(out_w, WTo, 256, 256);
    wconv_kernel<<<(256 * 1024 + 255) / 256, 256, 0, stream>>>(ff1_w, WTf1, 256, 1024);
    wconv_kernel<<<(1024 * 256 + 255) / 256, 256, 0, stream>>>(ff2_w, WTf2, 1024, 256);

    // embedding + positional encoding
    embed_kernel<<<(NROWS * DMODEL) / 256, 256, 0, stream>>>(x, in_w, in_b, h);

    for (int layer = 0; layer < 6; ++layer) {
        // --- attention sublayer ---
        ln_kernel<<<NROWS / 8, 256, 0, stream>>>(h, nullptr, ln1_g, ln1_b,
                                                 s2, s2b, nullptr);
        // tiles = (8192/32)*(768/32) = 6144 -> 768 blocks of 8 waves
        qkv_gemm_kernel<<<768, 256, 0, stream>>>(s2b, WTq, qkv_b,
                                                 qbuf, kbuf, vtb);
        attn_kernel<<<64 * 64, 128, 0, stream>>>(qbuf, kbuf, vtb, ctxb);
        // tiles = 256*8 = 2048 -> 256 blocks
        gemm_bf16_wave<<<256, 256, 0, stream>>>(
            ctxb, WTo, out_b, nullptr, oprj, nullptr, NROWS, DMODEL, DMODEL);
        // h += LN(o + s2)
        ln_kernel<<<NROWS / 8, 256, 0, stream>>>(oprj, s2, lna_g, lna_b,
                                                 nullptr, nullptr, h);

        // --- FFN sublayer ---
        ln_kernel<<<NROWS / 8, 256, 0, stream>>>(h, nullptr, ln2_g, ln2_b,
                                                 nullptr, s2b, nullptr);
        // tiles = 256*32 = 8192 -> 1024 blocks
        gemm_bf16_wave<<<1024, 256, 0, stream>>>(
            s2b, WTf1, ff1_b, nullptr, ff, nullptr, NROWS, DFF, DMODEL);
        vol_kernel<<<32, 256, 0, stream>>>(ff, ffb);
        // h = h + (f @ ff2 + b) ; tiles = 2048 -> 256 blocks
        gemm_bf16_wave<<<256, 256, 0, stream>>>(
            ffb, WTf2, ff2_b, h, h, nullptr, NROWS, DMODEL, DFF);
    }

    // final LayerNorm -> output
    ln_kernel<<<NROWS / 8, 256, 0, stream>>>(h, nullptr, lnf_g, lnf_b,
                                             (float*)d_out, nullptr, nullptr);
}
